// MultiHeadLatentAttention_69432441307203
// MI455X (gfx1250) — compile-verified, matrix operand order checked by reading, not run on hardware
//
#include <hip/hip_runtime.h>

// ---------------------------------------------------------------------------
// CDNA5 (gfx1250) Multi-Head Latent Attention, bf16 WMMA pipeline.
// Fixed problem size: B=8, S=4096, D=1024, H=16, dh=64, L=64.
// Design: one-shot f32->bf16 conversion of activations/weights, then all
// GEMMs are pure-bf16 with async global->LDS staging (ASYNCcnt) and WMMA.
// ---------------------------------------------------------------------------

typedef __attribute__((ext_vector_type(16))) __bf16 v16bf;
typedef __attribute__((ext_vector_type(8)))  float  v8f;

#if __has_builtin(__builtin_amdgcn_global_load_async_to_lds_b128) && \
    __has_builtin(__builtin_amdgcn_s_wait_asynccnt)
#define HAVE_ASYNC 1
// builtin signature: (int_v4 AS1* src, int_v4 AS3* dst, imm offset, imm cpol)
typedef int v4i_vec __attribute__((vector_size(16)));
#define ASYNC_GP(p) ((__attribute__((address_space(1))) v4i_vec*)(p))
#define ASYNC_LP(p) ((__attribute__((address_space(3))) v4i_vec*)(p))
#else
#define HAVE_ASYNC 0
#endif

__device__ __forceinline__ __bf16 f2bf(float f) {
  union { float f; unsigned u; } v; v.f = f;
  unsigned r = (v.u + 0x7FFFu + ((v.u >> 16) & 1u)) >> 16;   // RNE
  union { unsigned short s; __bf16 b; } o; o.s = (unsigned short)r;
  return o.b;
}

// pack two f32 -> packed bf16 pair (lo in [15:0]); prefer v_cvt_pk_bf16_f32
__device__ __forceinline__ unsigned bfpack(float lo, float hi) {
#if __has_builtin(__builtin_amdgcn_cvt_pk_bf16_f32)
  auto p = __builtin_amdgcn_cvt_pk_bf16_f32(lo, hi);
  union { decltype(p) v; unsigned u; } uu; uu.v = p;
  return uu.u;
#else
  union { float f; unsigned u; } a, b; a.f = lo; b.f = hi;
  unsigned ra = (a.u + 0x7FFFu + ((a.u >> 16) & 1u)) >> 16;
  unsigned rb = (b.u + 0x7FFFu + ((b.u >> 16) & 1u)) >> 16;
  return ra | (rb << 16);
#endif
}

// inv_freq = 10000^(-(2p)/64) = exp2(-log2(10000)/64 * (2p))
#define ROPE_C (-0.20762050593f)

// Load one 16x16x32 bf16 WMMA A/B fragment from an LDS tile stored row-major.
// Lane half h reads K=8h..8h+7 and K=16+8h..16+8h+7: exactly fragment
// elements 0-7 / 8-15 per the CDNA5 16-bit A/B VGPR layout (ISA 7.12.2).
__device__ __forceinline__ v16bf load_frag(const __bf16* rowp, int h) {
  union { v16bf v; uint4 q[2]; } u;
  u.q[0] = *reinterpret_cast<const uint4*>(rowp + 8 * h);
  u.q[1] = *reinterpret_cast<const uint4*>(rowp + 16 + 8 * h);
  return u.v;
}

#define WMMA_BF16(a, b, c) \
  __builtin_amdgcn_wmma_f32_16x16x32_bf16(false, (a), false, (b), (short)0, (c), false, false)

// ---------------------------------------------------------------------------
// Kernel 0: bulk f32 -> bf16 conversion, 8 elements/thread, exact-size grids.
// ---------------------------------------------------------------------------
__global__ __launch_bounds__(256) void cvt_bf16_kernel(
    const float* __restrict__ in, __bf16* __restrict__ out) {
  size_t i = ((size_t)blockIdx.x * 256 + threadIdx.x) * 8;
  float4 a = *(const float4*)(in + i);
  float4 b = *(const float4*)(in + i + 4);
  *(uint4*)(out + i) = make_uint4(bfpack(a.x, a.y), bfpack(a.z, a.w),
                                  bfpack(b.x, b.y), bfpack(b.z, b.w));
}

// ---------------------------------------------------------------------------
// Kernel 1: latent queries  lq[l,d] = latents[l,:]·q_w[d,:] + q_b[d],
// RoPE(pos=l), times 1/sqrt(dh)=0.125, stored bf16 as [h][l][dh].
// ---------------------------------------------------------------------------
__global__ __launch_bounds__(256) void lq_kernel(
    const float* __restrict__ latents, const float* __restrict__ qw,
    const float* __restrict__ qb, __bf16* __restrict__ lqo) {
  int idx = blockIdx.x * 256 + threadIdx.x;      // 64*1024 total
  int l = idx >> 10, d = idx & 1023;
  const float4* lr = (const float4*)(latents + (size_t)l * 1024);
  const float4* wr = (const float4*)(qw + (size_t)d * 1024);
  float acc = 0.f;
#pragma unroll 4
  for (int k = 0; k < 256; ++k) {
    float4 x = lr[k], y = wr[k];
    acc += x.x * y.x + x.y * y.y + x.z * y.z + x.w * y.w;
  }
  acc += qb[d];
  float other = __shfl_xor(acc, 1);              // interleaved-pair partner
  int din = d & 63;
  float inv_freq = __builtin_amdgcn_exp2f(ROPE_C * (float)(din & ~1));
  float sn, cs;
  __sincosf((float)l * inv_freq, &sn, &cs);
  float rv = (din & 1) ? (acc * cs + other * sn) : (acc * cs - other * sn);
  rv *= 0.125f;                                  // fold in 1/sqrt(dh)
  lqo[(((size_t)(d >> 6)) * 64 + l) * 64 + din] = f2bf(rv);
}

// ---------------------------------------------------------------------------
// Kernel 2/4: C[M,1024] = A[M,1024] @ W[1024,1024]^T + bias, all-bf16 inputs.
// 128x64 block tile, BK=32, 8 waves, each wave owns a 2x2 grid of 16x16
// subtiles (4 WMMAs per K-step). Double-buffered LDS with async global->LDS
// staging: zero VALU in the staging path, one s_wait_asynccnt + barrier per
// K-step, loads for step kt+1 overlapped with WMMA of step kt.
//   mode 0: K projection  -> RoPE(pos=s), bf16 store to k[b][h][s][d]
//   mode 1: V projection  -> bf16 store transposed to vT[b][h][d][s]
//   mode 2: out projection-> f32 store to outF[row][1024]
// ---------------------------------------------------------------------------
#define ASZ (128 * 40)
#define BSZ (64 * 40)
__global__ __launch_bounds__(256) void gemm_bf16_kernel(
    const __bf16* __restrict__ A, const __bf16* __restrict__ W,
    const float* __restrict__ bias, __bf16* __restrict__ outB,
    float* __restrict__ outF, int mode) {
  __shared__ __bf16 As[2 * ASZ];   // [buf][128 rows][stride 40]
  __shared__ __bf16 Bs[2 * BSZ];   // [buf][ 64 rows][stride 40]

  const int tid = threadIdx.x;
  const int w = tid >> 5, ln = tid & 31, lh = ln >> 4, lc = ln & 15;
  const int m0 = (w & 3) * 2;           // wave's 2 m-subtiles: m0, m0+1
  const int n0 = (w >> 2) * 2;          // wave's 2 n-subtiles: n0, n0+1
  const int mBase = blockIdx.x * 128, nBase = blockIdx.y * 64;

  // staging: A tile 128x32 -> 16 bf16/thread (2x b128), B 64x32 -> 8 (1x b128)
  const __bf16* ap = A + (size_t)(mBase + (tid >> 1)) * 1024 + (tid & 1) * 16;
  const __bf16* wp = W + (size_t)(nBase + (tid >> 2)) * 1024 + (tid & 3) * 8;
  __bf16* asd = &As[(tid >> 1) * 40 + (tid & 1) * 16];
  __bf16* bsd = &Bs[(tid >> 2) * 40 + (tid & 3) * 8];

  auto stage = [&](int kt, int p) {
#if HAVE_ASYNC
    __builtin_amdgcn_global_load_async_to_lds_b128(
        ASYNC_GP(ap + kt * 32),     ASYNC_LP(asd + p * ASZ), 0, 0);
    __builtin_amdgcn_global_load_async_to_lds_b128(
        ASYNC_GP(ap + kt * 32 + 8), ASYNC_LP(asd + p * ASZ + 8), 0, 0);
    __builtin_amdgcn_global_load_async_to_lds_b128(
        ASYNC_GP(wp + kt * 32),     ASYNC_LP(bsd + p * BSZ), 0, 0);
#else
    *(uint4*)(asd + p * ASZ)     = *(const uint4*)(ap + kt * 32);
    *(uint4*)(asd + p * ASZ + 8) = *(const uint4*)(ap + kt * 32 + 8);
    *(uint4*)(bsd + p * BSZ)     = *(const uint4*)(wp + kt * 32);
#endif
    __builtin_prefetch(ap + kt * 32 + 128, 0, 1);   // global_prefetch_b8
  };

  v8f acc00 = {0.f, 0.f, 0.f, 0.f, 0.f, 0.f, 0.f, 0.f};
  v8f acc01 = acc00, acc10 = acc00, acc11 = acc00;

  const int nk = 1024 / 32;
  stage(0, 0);
  for (int kt = 0; kt < nk; ++kt) {
#if HAVE_ASYNC
    __builtin_amdgcn_s_wait_asynccnt(0);           // tile kt landed in LDS
#endif
    __syncthreads();
    const int p = kt & 1;
    if (kt + 1 < nk) stage(kt + 1, (kt + 1) & 1);  // overlap with WMMA below
    const __bf16* Ab = &As[p * ASZ];
    const __bf16* Bb = &Bs[p * BSZ];
    v16bf af0 = load_frag(&Ab[(16 * (m0 + 0) + lc) * 40], lh);
    v16bf af1 = load_frag(&Ab[(16 * (m0 + 1) + lc) * 40], lh);
    v16bf bf0 = load_frag(&Bb[(16 * (n0 + 0) + lc) * 40], lh);
    v16bf bf1 = load_frag(&Bb[(16 * (n0 + 1) + lc) * 40], lh);
    acc00 = WMMA_BF16(af0, bf0, acc00);
    acc01 = WMMA_BF16(af0, bf1, acc01);
    acc10 = WMMA_BF16(af1, bf0, acc10);
    acc11 = WMMA_BF16(af1, bf1, acc11);
  }

#pragma unroll
  for (int mi = 0; mi < 2; ++mi) {
#pragma unroll
    for (int j = 0; j < 2; ++j) {
      v8f a = mi ? (j ? acc11 : acc10) : (j ? acc01 : acc00);
      int ng = nBase + 16 * (n0 + j) + lc;     // global output column
      float bv = bias[ng];
      if (mode == 2) {
#pragma unroll
        for (int r = 0; r < 8; ++r) {
          int rg = mBase + 16 * (m0 + mi) + r + 8 * lh;
          outF[(size_t)rg * 1024 + ng] = a[r] + bv;
        }
      } else {
        int din = ng & 63, hh = ng >> 6;
        float inv_freq = __builtin_amdgcn_exp2f(ROPE_C * (float)(din & ~1));
        int odd = din & 1;
#pragma unroll
        for (int r = 0; r < 8; ++r) {
          int rg = mBase + 16 * (m0 + mi) + r + 8 * lh; // row = b*4096 + s
          int bb = rg >> 12, s = rg & 4095;
          float v = a[r] + bv;
          float o = __shfl_xor(v, 1);                   // partner col ^ 1
          if (mode == 0) {                              // K: RoPE -> [b][h][s][d]
            float sn, cs;
            __sincosf((float)s * inv_freq, &sn, &cs);
            float rv = odd ? (v * cs + o * sn) : (v * cs - o * sn);
            outB[(((size_t)(bb * 16 + hh)) * 4096 + s) * 64 + din] = f2bf(rv);
          } else {                                      // V: -> vT[b][h][d][s]
            outB[(((size_t)(bb * 16 + hh)) * 64 + din) * 4096 + s] = f2bf(v);
          }
        }
      }
    }
  }
}

// ---------------------------------------------------------------------------
// Kernel 3: flash attention. One block per (b,h); 64 latent queries resident.
// Loop over S in 64-wide tiles; scores & P·V both via bf16 WMMA; online
// softmax row stats in LDS. Probs buffer aliases the K-tile buffer. Tile
// staging uses async global->LDS (ASYNCcnt). Output written in bf16 so the
// final projection GEMM consumes bf16 directly.
// ---------------------------------------------------------------------------
__global__ __launch_bounds__(256) void attn_kernel(
    const __bf16* __restrict__ lq, const __bf16* __restrict__ kb,
    const __bf16* __restrict__ vtb, __bf16* __restrict__ aob) {
  __shared__ __bf16 lqs[64 * 72];   // [l][d]
  __shared__ __bf16 kps[64 * 72];   // [s][d] K tile, reused as probs [l][s]
  __shared__ __bf16 vts[64 * 72];   // [d][s]
  __shared__ float  scs[64 * 68];   // scores [l][s]
  __shared__ float  mrow[64], lsum[64], arow[64];

  const int bh = blockIdx.x;
  const int b = bh >> 4, h = bh & 15;
  const int tid = threadIdx.x;
  const int w = tid >> 5, ln = tid & 31, lh2 = ln >> 4, lc = ln & 15;
  const int m_sub = w & 3, n0 = (w >> 2) * 2;
  const int lrow = tid >> 2, lcb = (tid & 3) * 16;

  { // stage latent queries once
    const __bf16* src = lq + ((size_t)(h * 64 + lrow)) * 64 + lcb;
    *(uint4*)&lqs[lrow * 72 + lcb]     = *(const uint4*)src;
    *(uint4*)&lqs[lrow * 72 + lcb + 8] = *(const uint4*)(src + 8);
  }
  if (tid < 64) { mrow[tid] = -3.0e38f; lsum[tid] = 0.f; }

  v8f oa0 = {0.f, 0.f, 0.f, 0.f, 0.f, 0.f, 0.f, 0.f};
  v8f oa1 = oa0;

  const __bf16* kbase  = kb  + (size_t)bh * 4096 * 64;
  const __bf16* vtbase = vtb + (size_t)bh * 64 * 4096;
  __syncthreads();

  for (int st = 0; st < 4096; st += 64) {
    { // stage K tile [s][d] and V^T tile [d][s]
      const __bf16* ks = kbase + (size_t)(st + lrow) * 64 + lcb;
      const __bf16* vs = vtbase + (size_t)lrow * 4096 + st + lcb;
#if HAVE_ASYNC
      __builtin_amdgcn_global_load_async_to_lds_b128(
          ASYNC_GP(ks),     ASYNC_LP(&kps[lrow * 72 + lcb]), 0, 0);
      __builtin_amdgcn_global_load_async_to_lds_b128(
          ASYNC_GP(ks + 8), ASYNC_LP(&kps[lrow * 72 + lcb + 8]), 0, 0);
      __builtin_amdgcn_global_load_async_to_lds_b128(
          ASYNC_GP(vs),     ASYNC_LP(&vts[lrow * 72 + lcb]), 0, 0);
      __builtin_amdgcn_global_load_async_to_lds_b128(
          ASYNC_GP(vs + 8), ASYNC_LP(&vts[lrow * 72 + lcb + 8]), 0, 0);
      __builtin_amdgcn_s_wait_asynccnt(0);
#else
      *(uint4*)&kps[lrow * 72 + lcb]     = *(const uint4*)ks;
      *(uint4*)&kps[lrow * 72 + lcb + 8] = *(const uint4*)(ks + 8);
      *(uint4*)&vts[lrow * 72 + lcb]     = *(const uint4*)vs;
      *(uint4*)&vts[lrow * 72 + lcb + 8] = *(const uint4*)(vs + 8);
#endif
    }
    __syncthreads();

    // scores[l, s] = lq_rope_scaled · k_rope   (dh=64 -> 2 WMMA K-steps)
    v8f sc0 = {0.f, 0.f, 0.f, 0.f, 0.f, 0.f, 0.f, 0.f};
    v8f sc1 = sc0;
#pragma unroll
    for (int kk = 0; kk < 2; ++kk) {
      v16bf af  = load_frag(&lqs[(16 * m_sub + lc) * 72 + 32 * kk], lh2);
      v16bf bf0 = load_frag(&kps[(16 * (n0 + 0) + lc) * 72 + 32 * kk], lh2);
      v16bf bf1 = load_frag(&kps[(16 * (n0 + 1) + lc) * 72 + 32 * kk], lh2);
      sc0 = WMMA_BF16(af, bf0, sc0);
      sc1 = WMMA_BF16(af, bf1, sc1);
    }
#pragma unroll
    for (int r = 0; r < 8; ++r) {
      int rr = 16 * m_sub + r + 8 * lh2;
      scs[rr * 68 + 16 * (n0 + 0) + lc] = sc0[r];
      scs[rr * 68 + 16 * (n0 + 1) + lc] = sc1[r];
    }
    __syncthreads();

    { // online softmax: 4 lanes per row (quad within a wave)
      int row = tid >> 2, q = tid & 3;
      float mo = mrow[row];
      float vals[16], mx = -3.0e38f;
#pragma unroll
      for (int i = 0; i < 16; ++i) {
        vals[i] = scs[row * 68 + q * 16 + i];
        mx = fmaxf(mx, vals[i]);
      }
      mx = fmaxf(mx, __shfl_xor(mx, 1));
      mx = fmaxf(mx, __shfl_xor(mx, 2));
      float mn = fmaxf(mo, mx);
      float al = __expf(mo - mn);
      float sum = 0.f;
      unsigned* pp = (unsigned*)&kps[row * 72 + q * 16]; // probs over K tile
#pragma unroll
      for (int i = 0; i < 8; ++i) {
        float p0 = __expf(vals[2 * i] - mn);
        float p1 = __expf(vals[2 * i + 1] - mn);
        sum += p0 + p1;
        pp[i] = bfpack(p0, p1);
      }
      sum += __shfl_xor(sum, 1);
      sum += __shfl_xor(sum, 2);
      if (q == 0) { mrow[row] = mn; lsum[row] = lsum[row] * al + sum; arow[row] = al; }
    }
    __syncthreads();

    // rescale accumulators, then O += P @ V  (K=64 over s -> 2 WMMA steps)
#pragma unroll
    for (int r = 0; r < 8; ++r) {
      float al = arow[16 * m_sub + r + 8 * lh2];
      oa0[r] *= al;
      oa1[r] *= al;
    }
#pragma unroll
    for (int kk = 0; kk < 2; ++kk) {
      v16bf af  = load_frag(&kps[(16 * m_sub + lc) * 72 + 32 * kk], lh2);
      v16bf bf0 = load_frag(&vts[(16 * (n0 + 0) + lc) * 72 + 32 * kk], lh2);
      v16bf bf1 = load_frag(&vts[(16 * (n0 + 1) + lc) * 72 + 32 * kk], lh2);
      oa0 = WMMA_BF16(af, bf0, oa0);
      oa1 = WMMA_BF16(af, bf1, oa1);
    }
    __syncthreads();
  }

  // final normalize; bf16 scatter to attn_out [b][l][h*64 + din]
#pragma unroll
  for (int j = 0; j < 2; ++j) {
    v8f a = j ? oa1 : oa0;
    int cg = 16 * (n0 + j) + lc;
#pragma unroll
    for (int r = 0; r < 8; ++r) {
      int rg = 16 * m_sub + r + 8 * lh2;
      float inv = 1.f / lsum[rg];
      aob[((size_t)(b * 64 + rg)) * 1024 + h * 64 + cg] = f2bf(a[r] * inv);
    }
  }
}

// ---------------------------------------------------------------------------
extern "C" void kernel_launch(void* const* d_in, const int* in_sizes, int n_in,
                              void* d_out, int out_size, void* d_ws, size_t ws_size,
                              hipStream_t stream) {
  const float* x       = (const float*)d_in[0];
  const float* latents = (const float*)d_in[1];
  const float* q_w     = (const float*)d_in[2];
  const float* q_b     = (const float*)d_in[3];
  const float* k_w     = (const float*)d_in[4];
  const float* k_b     = (const float*)d_in[5];
  const float* v_w     = (const float*)d_in[6];
  const float* v_b     = (const float*)d_in[7];
  const float* out_w   = (const float*)d_in[8];
  const float* out_b   = (const float*)d_in[9];
  float* out = (float*)d_out;

  const int B = 8, S = 4096, D = 1024, H = 16, L = 64;
  char* ws = (char*)d_ws;
  size_t off = 0;
  __bf16* xbf   = (__bf16*)(ws + off); off += (size_t)B * S * D * 2;      // 64 MB
  __bf16* kbuf  = (__bf16*)(ws + off); off += (size_t)B * H * S * 64 * 2; // 64 MB
  __bf16* vtbuf = (__bf16*)(ws + off); off += (size_t)B * H * 64 * S * 2; // 64 MB
  __bf16* kwbf  = (__bf16*)(ws + off); off += (size_t)D * D * 2;          //  2 MB
  __bf16* vwbf  = (__bf16*)(ws + off); off += (size_t)D * D * 2;          //  2 MB
  __bf16* owbf  = (__bf16*)(ws + off); off += (size_t)D * D * 2;          //  2 MB
  __bf16* lqbuf = (__bf16*)(ws + off); off += (size_t)H * L * 64 * 2;     // 128 KB
  __bf16* aobuf = (__bf16*)(ws + off); off += (size_t)B * L * D * 2;      //  1 MB
  (void)ws_size; (void)in_sizes; (void)n_in; (void)out_size;

  // 0) one-shot f32 -> bf16 conversions (bandwidth-bound, ~9 us total)
  cvt_bf16_kernel<<<(B * S * D) / 2048, 256, 0, stream>>>(x, xbf);
  cvt_bf16_kernel<<<(D * D) / 2048, 256, 0, stream>>>(k_w, kwbf);
  cvt_bf16_kernel<<<(D * D) / 2048, 256, 0, stream>>>(v_w, vwbf);
  cvt_bf16_kernel<<<(D * D) / 2048, 256, 0, stream>>>(out_w, owbf);

  // 1) latent queries (RoPE + 1/sqrt(dh) folded in)
  lq_kernel<<<(L * D) / 256, 256, 0, stream>>>(latents, q_w, q_b, lqbuf);

  // 2) K and V projections (the FLOP-dominant GEMMs)
  dim3 gproj((B * S) / 128, D / 64);
  gemm_bf16_kernel<<<gproj, 256, 0, stream>>>(xbf, kwbf, k_b, kbuf, nullptr, 0);
  gemm_bf16_kernel<<<gproj, 256, 0, stream>>>(xbf, vwbf, v_b, vtbuf, nullptr, 1);

  // 3) flash attention, one block per (b,h)
  attn_kernel<<<B * H, 256, 0, stream>>>(lqbuf, kbuf, vtbuf, aobuf);

  // 4) output projection (bf16 in, f32 out)
  dim3 gout((B * L) / 128, D / 64);
  gemm_bf16_kernel<<<gout, 256, 0, stream>>>(aobuf, owbf, out_b, nullptr, out, 2);
}
